// MultiHeadLatentAttention_70626442215501
// MI455X (gfx1250) — matrix operand order repaired and checked by reading, not compiled, not run on hardware
//
#include <hip/hip_runtime.h>

// ---------------- problem constants ----------------
constexpr int kT    = 2048;   // sequence length
constexpr int kDim  = 2048;
constexpr int kNH   = 16;
constexpr int kHD   = 128;
constexpr int kQC   = 1536;
constexpr int kKVC  = 512;
constexpr int kDR   = 64;
constexpr int kHDR  = kHD + kDR;     // 192
constexpr int kQK   = kNH * kHDR;    // 3072 (concat q/k row width)
constexpr int kNHD  = kNH * kHD;     // 2048
constexpr int kNDR  = kNH * kDR;     // 1024

typedef __bf16 bf16;
typedef __attribute__((ext_vector_type(8)))  bf16  v8bf;
typedef __attribute__((ext_vector_type(16))) bf16  v16bf;
typedef __attribute__((ext_vector_type(8)))  float v8f;

__device__ __forceinline__ bf16 f2bf(float f) {
  unsigned u = __builtin_bit_cast(unsigned, f);
  unsigned r = u + 0x7FFFu + ((u >> 16) & 1u);   // round-to-nearest-even
  unsigned short h = (unsigned short)(r >> 16);
  return __builtin_bit_cast(bf16, h);
}

// ---------------- fp32 -> bf16 convert (row-major copy) ----------------
__global__ void cvt_f32_bf16(const float* __restrict__ in, bf16* __restrict__ out, int n) {
  int i = blockIdx.x * blockDim.x + threadIdx.x;
  if (i < n) out[i] = f2bf(in[i]);
}

// ---------------- fp32 [R,C] -> bf16 transposed [C,R], LDS-tiled ----------------
__global__ void cvt_transpose_bf16(const float* __restrict__ in, bf16* __restrict__ out,
                                   int R, int C) {
  __shared__ float tile[32][33];          // +1 pad: bank-conflict-free transpose
  const int c0 = blockIdx.x * 32;
  const int r0 = blockIdx.y * 32;
  const int tx = threadIdx.x;             // 0..31
  for (int i = threadIdx.y; i < 32; i += 8)
    tile[i][tx] = in[(size_t)(r0 + i) * C + c0 + tx];
  __syncthreads();
  for (int i = threadIdx.y; i < 32; i += 8)
    out[(size_t)(c0 + i) * R + r0 + tx] = f2bf(tile[tx][i]);
}

// ---------------- RMSNorm (block per row) ----------------
__global__ void rmsnorm_kernel(const float* __restrict__ X, const float* __restrict__ g,
                               bf16* __restrict__ Y, int C) {
  __shared__ float red[256];
  const int row = blockIdx.x;
  const float* x = X + (size_t)row * C;
  bf16* y = Y + (size_t)row * C;
  float ss = 0.f;
  for (int j = threadIdx.x; j < C; j += 256) { float v = x[j]; ss += v * v; }
  red[threadIdx.x] = ss; __syncthreads();
  for (int s = 128; s > 0; s >>= 1) {
    if (threadIdx.x < s) red[threadIdx.x] += red[threadIdx.x + s];
    __syncthreads();
  }
  const float r = rsqrtf(red[0] / (float)C + 1e-6f);
  for (int j = threadIdx.x; j < C; j += 256) y[j] = f2bf(x[j] * r * g[j]);
}

// ---------------- RoPE: fill rope slots of q_full / k_full ----------------
__global__ void rope_fill(const float* __restrict__ qr, const float* __restrict__ kr,
                          const float* __restrict__ cosT, const float* __restrict__ sinT,
                          bf16* __restrict__ qf, bf16* __restrict__ kf) {
  const int t = blockIdx.x;
  const int tid = threadIdx.x;            // 512 = 16 heads * 32 pairs
  const int h = tid >> 5, i = tid & 31;
  const float c = cosT[t * (kDR / 2) + i];
  const float s = sinT[t * (kDR / 2) + i];
  const float x0 = qr[(size_t)t * kNDR + h * kDR + 2 * i];
  const float x1 = qr[(size_t)t * kNDR + h * kDR + 2 * i + 1];
  const size_t qb = (size_t)t * kQK + h * kHDR + kHD + 2 * i;
  qf[qb]     = f2bf(x0 * c - x1 * s);
  qf[qb + 1] = f2bf(x0 * s + x1 * c);
  if (h == 0) {
    const float k0 = kr[(size_t)t * kDR + 2 * i];
    const float k1 = kr[(size_t)t * kDR + 2 * i + 1];
    const float o0 = k0 * c - k1 * s;
    const float o1 = k0 * s + k1 * c;
    for (int hh = 0; hh < kNH; ++hh) {
      const size_t kb = (size_t)t * kQK + hh * kHDR + kHD + 2 * i;
      kf[kb] = f2bf(o0); kf[kb + 1] = f2bf(o1);
    }
  }
}

// ---------------- row softmax -> bf16 ----------------
__global__ void softmax_rows(const float* __restrict__ S, bf16* __restrict__ P) {
  __shared__ float red[256];
  const int row = blockIdx.x;
  const float* s = S + (size_t)row * kT;
  bf16* p = P + (size_t)row * kT;
  float mx = -3.4e38f;
  for (int j = threadIdx.x; j < kT; j += 256) mx = fmaxf(mx, s[j]);
  red[threadIdx.x] = mx; __syncthreads();
  for (int st = 128; st > 0; st >>= 1) {
    if (threadIdx.x < st) red[threadIdx.x] = fmaxf(red[threadIdx.x], red[threadIdx.x + st]);
    __syncthreads();
  }
  mx = red[0]; __syncthreads();
  float sum = 0.f;
  for (int j = threadIdx.x; j < kT; j += 256) sum += __expf(s[j] - mx);
  red[threadIdx.x] = sum; __syncthreads();
  for (int st = 128; st > 0; st >>= 1) {
    if (threadIdx.x < st) red[threadIdx.x] += red[threadIdx.x + st];
    __syncthreads();
  }
  const float inv = 1.0f / red[0];
  for (int j = threadIdx.x; j < kT; j += 256) p[j] = f2bf(__expf(s[j] - mx) * inv);
}

// ---------------- NT WMMA GEMM: C = A @ B^T ----------------
// A: bf16 row-major [M,K] (lda).  B: bf16 row-major [N,K] (ldb) -> all K-contiguous loads.
// Each wave computes a 32x32 tile (2x2 WMMA 16x16 tiles); 4 waves/block cover 64x64.
// EPI: 0 f32 out | 1 bf16 out | 2 bf16 head-remap (n -> (n/128)*192 + n%128)
//      3 f32 scores: acc*scale + mask[row*kT+n] | 4 bf16 transposed out C^T[n*ldc+row]
constexpr int EPI_F32 = 0, EPI_BF16 = 1, EPI_HEADS = 2, EPI_SCORES = 3, EPI_BF16T = 4;

__device__ __forceinline__ v16bf ld_fragA(const bf16* p) {
  // lane<16: K {0..7,16..23}; lane>=16: K {8..15,24..31} (caller applied +0/+8)
  v8bf lo = *(const v8bf*)(p);
  v8bf hi = *(const v8bf*)(p + 16);
  return __builtin_shufflevector(lo, hi, 0,1,2,3,4,5,6,7,8,9,10,11,12,13,14,15);
}
__device__ __forceinline__ v16bf ld_fragB(const bf16* p) {
  // lane<16: K {0..7,16..23}; lane>=16: K {8..15,24..31} (caller applied +0/+8)
  v8bf lo = *(const v8bf*)(p);
  v8bf hi = *(const v8bf*)(p + 16);
  return __builtin_shufflevector(lo, hi, 0,1,2,3,4,5,6,7,8,9,10,11,12,13,14,15);
}

template <int EPI>
__global__ void gemm_nt_wmma(const bf16* __restrict__ A, int lda,
                             const bf16* __restrict__ B, int ldb,
                             void* __restrict__ Cout, int ldc,
                             int M, int N, int K,
                             const float* __restrict__ mask, float scale) {
  const int lane  = threadIdx.x;            // 0..31
  const int waveN = threadIdx.y & 1;
  const int waveM = threadIdx.y >> 1;
  const int m0 = blockIdx.y * 64 + waveM * 32;
  const int n0 = blockIdx.x * 64 + waveN * 32;
  if (m0 >= M || n0 >= N) return;

  const int l15   = lane & 15;
  const int koffA = (lane < 16) ? 0 : 8;
  const int koffB = (lane < 16) ? 0 : 8;

  const bf16* a0row = A + (size_t)(m0 + l15) * lda + koffA;
  const bf16* a1row = A + (size_t)(m0 + 16 + l15) * lda + koffA;
  const bf16* b0row = B + (size_t)(n0 + l15) * ldb + koffB;
  const bf16* b1row = B + (size_t)(n0 + 16 + l15) * ldb + koffB;

  v8f acc00 = {}, acc01 = {}, acc10 = {}, acc11 = {};

  for (int k0 = 0; k0 < K; k0 += 32) {
    if (k0 + 128 < K) {
      __builtin_prefetch(a0row + k0 + 128, 0, 1);
      __builtin_prefetch(a1row + k0 + 128, 0, 1);
      __builtin_prefetch(b0row + k0 + 128, 0, 1);
      __builtin_prefetch(b1row + k0 + 128, 0, 1);
    }
    v16bf a0 = ld_fragA(a0row + k0);
    v16bf a1 = ld_fragA(a1row + k0);
    v16bf b0 = ld_fragB(b0row + k0);
    v16bf b1 = ld_fragB(b1row + k0);
    acc00 = __builtin_amdgcn_wmma_f32_16x16x32_bf16(false, a0, false, b0, (short)0, acc00, false, false);
    acc01 = __builtin_amdgcn_wmma_f32_16x16x32_bf16(false, a0, false, b1, (short)0, acc01, false, false);
    acc10 = __builtin_amdgcn_wmma_f32_16x16x32_bf16(false, a1, false, b0, (short)0, acc10, false, false);
    acc11 = __builtin_amdgcn_wmma_f32_16x16x32_bf16(false, a1, false, b1, (short)0, acc11, false, false);
  }

  // C/D layout: VGPR r -> M = r (lanes 0-15) / r+8 (lanes 16-31); N = lane%16
  const int mhalf = (lane < 16) ? 0 : 8;
  #pragma unroll
  for (int mi = 0; mi < 2; ++mi) {
    #pragma unroll
    for (int ni = 0; ni < 2; ++ni) {
      const v8f acc = (mi == 0) ? (ni == 0 ? acc00 : acc01)
                                : (ni == 0 ? acc10 : acc11);
      const int n     = n0 + ni * 16 + l15;
      const int mbase = m0 + mi * 16 + mhalf;
      #pragma unroll
      for (int r = 0; r < 8; ++r) {
        const int row = mbase + r;
        const float v = acc[r];
        if (EPI == EPI_F32) {
          ((float*)Cout)[(size_t)row * ldc + n] = v;
        } else if (EPI == EPI_BF16) {
          ((bf16*)Cout)[(size_t)row * ldc + n] = f2bf(v);
        } else if (EPI == EPI_HEADS) {
          const int col = (n / kHD) * kHDR + (n % kHD);
          ((bf16*)Cout)[(size_t)row * ldc + col] = f2bf(v);
        } else if (EPI == EPI_SCORES) {
          const float mv = mask[(size_t)row * kT + n];
          ((float*)Cout)[(size_t)row * ldc + n] = v * scale + mv;
        } else { // EPI_BF16T: store C^T
          ((bf16*)Cout)[(size_t)n * ldc + row] = f2bf(v);
        }
      }
    }
  }
}

// ---------------- host-side orchestration ----------------
static inline dim3 gemm_grid(int M, int N) { return dim3(N / 64, M / 64); }

extern "C" void kernel_launch(void* const* d_in, const int* in_sizes, int n_in,
                              void* d_out, int out_size, void* d_ws, size_t ws_size,
                              hipStream_t stream) {
  const float* x     = (const float*)d_in[0];
  const float* W_dq  = (const float*)d_in[1];
  const float* W_uq  = (const float*)d_in[2];
  const float* W_dkv = (const float*)d_in[3];
  const float* W_uk  = (const float*)d_in[4];
  const float* W_uv  = (const float*)d_in[5];
  const float* W_qr  = (const float*)d_in[6];
  const float* W_kr  = (const float*)d_in[7];
  const float* W_o   = (const float*)d_in[8];
  const float* g_q   = (const float*)d_in[9];
  const float* g_kv  = (const float*)d_in[10];
  const float* fcos  = (const float*)d_in[11];
  const float* fsin  = (const float*)d_in[12];
  const float* mask  = (const float*)d_in[13];
  float* out = (float*)d_out;

  char* ws = (char*)d_ws;
  size_t off = 0;
  auto take = [&](size_t bytes) -> char* {
    char* p = ws + off;
    off += (bytes + 255) & ~(size_t)255;
    return p;
  };

  bf16* xb     = (bf16*)take((size_t)kT * kDim * 2);
  // transposed bf16 weights: Wt[n][k] = W[k][n]
  bf16* wdq_t  = (bf16*)take((size_t)kQC * kDim * 2);
  bf16* wdkv_t = (bf16*)take((size_t)kKVC * kDim * 2);
  bf16* wuq_t  = (bf16*)take((size_t)kNHD * kQC * 2);
  bf16* wuk_t  = (bf16*)take((size_t)kNHD * kKVC * 2);
  bf16* wuv_t  = (bf16*)take((size_t)kNHD * kKVC * 2);
  bf16* wqr_t  = (bf16*)take((size_t)kNDR * kQC * 2);
  bf16* wkr_t  = (bf16*)take((size_t)kDR * kDim * 2);
  bf16* wo_t   = (bf16*)take((size_t)kDim * kNHD * 2);
  float* qc_raw  = (float*)take((size_t)kT * kQC * 4);
  float* kvc_raw = (float*)take((size_t)kT * kKVC * 4);
  bf16* qc_b   = (bf16*)take((size_t)kT * kQC * 2);
  bf16* kvc_b  = (bf16*)take((size_t)kT * kKVC * 2);
  float* qr_raw = (float*)take((size_t)kT * kNDR * 4);
  float* kr_raw = (float*)take((size_t)kT * kDR * 4);
  bf16* q_full = (bf16*)take((size_t)kT * kQK * 2);
  bf16* k_full = (bf16*)take((size_t)kT * kQK * 2);
  bf16* vT     = (bf16*)take((size_t)kNHD * kT * 2);   // v transposed: [n, t]
  float* scores = (float*)take((size_t)kT * kT * 4);
  bf16* attn_b = (bf16*)take((size_t)kT * kT * 2);
  bf16* o_heads = (bf16*)take((size_t)kT * kNHD * 2);

  const dim3 blk(32, 4);
  const dim3 tblk(32, 8);
  const float scale = 1.0f / sqrtf((float)kHDR);

  // 1) convert x; convert+transpose all weights
  cvt_f32_bf16<<<(unsigned)(((size_t)kT * kDim + 255) / 256), 256, 0, stream>>>(
      x, xb, kT * kDim);
  auto tconv = [&](const float* src, bf16* dst, int R, int C) {
    cvt_transpose_bf16<<<dim3(C / 32, R / 32), tblk, 0, stream>>>(src, dst, R, C);
  };
  tconv(W_dq,  wdq_t,  kDim, kQC);
  tconv(W_dkv, wdkv_t, kDim, kKVC);
  tconv(W_uq,  wuq_t,  kQC,  kNHD);
  tconv(W_uk,  wuk_t,  kKVC, kNHD);
  tconv(W_uv,  wuv_t,  kKVC, kNHD);
  tconv(W_qr,  wqr_t,  kQC,  kNDR);
  tconv(W_kr,  wkr_t,  kDim, kDR);
  tconv(W_o,   wo_t,   kNHD, kDim);

  // 2) down-projections
  gemm_nt_wmma<EPI_F32><<<gemm_grid(kT, kQC), blk, 0, stream>>>(
      xb, kDim, wdq_t, kDim, qc_raw, kQC, kT, kQC, kDim, nullptr, 0.f);
  gemm_nt_wmma<EPI_F32><<<gemm_grid(kT, kKVC), blk, 0, stream>>>(
      xb, kDim, wdkv_t, kDim, kvc_raw, kKVC, kT, kKVC, kDim, nullptr, 0.f);

  // 3) rmsnorms
  rmsnorm_kernel<<<kT, 256, 0, stream>>>(qc_raw, g_q, qc_b, kQC);
  rmsnorm_kernel<<<kT, 256, 0, stream>>>(kvc_raw, g_kv, kvc_b, kKVC);

  // 4) up-projections (q/k no-rope parts written straight into concat layout)
  gemm_nt_wmma<EPI_HEADS><<<gemm_grid(kT, kNHD), blk, 0, stream>>>(
      qc_b, kQC, wuq_t, kQC, q_full, kQK, kT, kNHD, kQC, nullptr, 0.f);
  gemm_nt_wmma<EPI_F32><<<gemm_grid(kT, kNDR), blk, 0, stream>>>(
      qc_b, kQC, wqr_t, kQC, qr_raw, kNDR, kT, kNDR, kQC, nullptr, 0.f);
  gemm_nt_wmma<EPI_HEADS><<<gemm_grid(kT, kNHD), blk, 0, stream>>>(
      kvc_b, kKVC, wuk_t, kKVC, k_full, kQK, kT, kNHD, kKVC, nullptr, 0.f);
  gemm_nt_wmma<EPI_BF16T><<<gemm_grid(kT, kNHD), blk, 0, stream>>>(
      kvc_b, kKVC, wuv_t, kKVC, vT, kT, kT, kNHD, kKVC, nullptr, 0.f);
  gemm_nt_wmma<EPI_F32><<<gemm_grid(kT, kDR), blk, 0, stream>>>(
      xb, kDim, wkr_t, kDim, kr_raw, kDR, kT, kDR, kDim, nullptr, 0.f);

  // 5) rope fill of the 64-wide rope slots
  rope_fill<<<kT, 512, 0, stream>>>(qr_raw, kr_raw, fcos, fsin, q_full, k_full);

  // 6) per-head attention
  for (int h = 0; h < kNH; ++h) {
    // scores = (q_h @ k_h^T) * scale + mask   (both operands K-contiguous)
    gemm_nt_wmma<EPI_SCORES><<<gemm_grid(kT, kT), blk, 0, stream>>>(
        q_full + h * kHDR, kQK, k_full + h * kHDR, kQK,
        scores, kT, kT, kT, kHDR, mask, scale);
    softmax_rows<<<kT, 256, 0, stream>>>(scores, attn_b);
    // o_h = attn @ v_h  (v stored transposed -> NT with contiguous K)
    gemm_nt_wmma<EPI_BF16><<<gemm_grid(kT, kHD), blk, 0, stream>>>(
        attn_b, kT, vT + (size_t)h * kHD * kT, kT,
        o_heads + h * kHD, kNHD, kT, kHD, kT, nullptr, 0.f);
  }

  // 7) output projection -> fp32 d_out
  gemm_nt_wmma<EPI_F32><<<gemm_grid(kT, kDim), blk, 0, stream>>>(
      o_heads, kNHD, wo_t, kNHD, out, kDim, kT, kDim, kNHD, nullptr, 0.f);
}
